// ContextAttentionDiffInput_927712936393
// MI455X (gfx1250) — matrix operand order repaired and checked by reading, not compiled, not run on hardware
//
#include <hip/hip_runtime.h>
#include <hip/hip_bf16.h>

// MI455X / gfx1250: wave32, WMMA bf16 16x16x32, f32 accumulate.
// All GEMM operands are laid out so every WMMA fragment is two global b128
// loads (A row-major K-contiguous; B stored column-major = Bt[N][K]).

typedef __attribute__((ext_vector_type(16))) __bf16 v16bf;
typedef __attribute__((ext_vector_type(8)))  __bf16 v8bf;
typedef __attribute__((ext_vector_type(8)))  float  v8f;

// ---------------------------------------------------------------------------
// elementwise f32 -> bf16 (weights / Qw)
// ---------------------------------------------------------------------------
__global__ void __launch_bounds__(256)
cvt_f32_bf16(const float* __restrict__ x, __bf16* __restrict__ y, int n) {
    int i = blockIdx.x * 256 + threadIdx.x;
    if (i < n) y[i] = (__bf16)x[i];
}

// ---------------------------------------------------------------------------
// LDS tile transpose: x [rows][cols] f32  ->  y [cols][rows] bf16  (per z)
// 16x16 tiles, coalesced read and write.
// ---------------------------------------------------------------------------
__global__ void __launch_bounds__(256)
transpose_f32_bf16(const float* __restrict__ x, __bf16* __restrict__ y,
                   int rows, int cols) {
    __shared__ float tile[16][17];
    const int    tx = threadIdx.x & 15, ty = threadIdx.x >> 4;
    const int    c0 = blockIdx.x * 16, r0 = blockIdx.y * 16;
    const size_t plane = (size_t)rows * cols * blockIdx.z;
    tile[ty][tx] = x[plane + (size_t)(r0 + ty) * cols + c0 + tx];
    __syncthreads();
    y[plane + (size_t)(c0 + ty) * rows + r0 + tx] = (__bf16)tile[tx][ty];
}

// ---------------------------------------------------------------------------
// Fragment loaders (two b128 each), per CDNA5 ISA 7.12.2 layouts.
// A 16x32: lanes 0-15 (M=lane): elems0-7=K0..7, 8-15=K16..23
//          lanes16-31 (M=lane-16): elems0-7=K8..15, 8-15=K24..31
// B 32x16 via Bt[N][K]: lane n&15 = column, same K layout as A
// ---------------------------------------------------------------------------
__device__ __forceinline__ v16bf load_a_frag(const __bf16* Atile, int lda) {
    const int lane = threadIdx.x & 31;
    const __bf16* p = Atile + (size_t)(lane & 15) * lda + (lane >> 4) * 8;
    v8bf lo = *(const v8bf*)(p);
    v8bf hi = *(const v8bf*)(p + 16);
    v16bf a;
#pragma unroll
    for (int i = 0; i < 8; ++i) { a[i] = lo[i]; a[i + 8] = hi[i]; }
    return a;
}

__device__ __forceinline__ v16bf load_b_frag(const __bf16* Bttile, int ldb) {
    const int lane = threadIdx.x & 31;
    const __bf16* p = Bttile + (size_t)(lane & 15) * ldb + (lane >> 4) * 8;
    v8bf lo = *(const v8bf*)(p);
    v8bf hi = *(const v8bf*)(p + 16);
    v16bf b;
#pragma unroll
    for (int i = 0; i < 8; ++i) { b[i] = lo[i]; b[i + 8] = hi[i]; }
    return b;
}

__device__ __forceinline__ v8f wmma_bf16(v16bf a, v16bf b, v8f c) {
    return __builtin_amdgcn_wmma_f32_16x16x32_bf16(false, a, false, b,
                                                   (short)0, c, false, false);
}

// ---------------------------------------------------------------------------
// bf16 WMMA GEMM: each wave owns a 32(M) x 64(N) strip = 8 accumulators,
// 4 waves / block. A: MxK row-major. Bt: NxK row-major (column-major B).
// Epilogue modes:
//   0: bf16 out[m][n] (ldo), ELU(acc + bias[m])   (V projections -> [d][l])
//   1: bf16 out[n][m] (ldo), ELU(acc + bias[m])   (K projections -> [l][d])
//   2: f32  out[m][n] (ldo)                       (E scores / final C2)
//   3: bf16 out[m][n] (ldo)                       (C1 context)
// ---------------------------------------------------------------------------
__global__ void __launch_bounds__(128)
gemm_wmma_bf16(const __bf16* __restrict__ A, const __bf16* __restrict__ Bt,
               int tilesN, int K, int lda, int ldb,
               void* __restrict__ out, int ldo,
               const float* __restrict__ bias, int mode) {
    const int lane = threadIdx.x & 31;
    const int w    = threadIdx.x >> 5;
    const int tile = blockIdx.x * 4 + w;
    const int tm   = tile / tilesN;           // 32-row strip
    const int tn   = tile % tilesN;           // 64-col strip
    const int n    = lane & 15;
    const int hv   = lane >> 4;

    const __bf16* A0 = A + (size_t)(tm * 32) * lda;
    const __bf16* A1 = A0 + (size_t)16 * lda;
    const __bf16* Bb = Bt + (size_t)(tn * 64) * ldb;

    v8f acc[8] = {};
    for (int k0 = 0; k0 < K; k0 += 32) {
        v16bf a0 = load_a_frag(A0 + k0, lda);
        v16bf a1 = load_a_frag(A1 + k0, lda);
#pragma unroll
        for (int t = 0; t < 4; ++t) {
            v16bf b = load_b_frag(Bb + (size_t)(t * 16) * ldb + k0, ldb);
            acc[t * 2]     = wmma_bf16(a0, b, acc[t * 2]);
            acc[t * 2 + 1] = wmma_bf16(a1, b, acc[t * 2 + 1]);
        }
    }

    // C/D layout: VGPR v -> M = v + 8*hv, N = lane&15
#pragma unroll
    for (int r = 0; r < 2; ++r) {
#pragma unroll
        for (int t = 0; t < 4; ++t) {
            const v8f& a = acc[t * 2 + r];
            const int mBase = tm * 32 + r * 16 + hv * 8;
            const int nCol  = tn * 64 + t * 16 + n;
            if (mode <= 1) {
                __bf16* O = (__bf16*)out;
#pragma unroll
                for (int v = 0; v < 8; ++v) {
                    const int m = mBase + v;
                    float x = a[v] + bias[m];
                    x = (x > 0.0f) ? x : (__expf(x) - 1.0f);   // ELU
                    if (mode == 0) O[(size_t)m    * ldo + nCol] = (__bf16)x;
                    else           O[(size_t)nCol * ldo + m   ] = (__bf16)x;
                }
            } else if (mode == 2) {
                float* O = (float*)out;
#pragma unroll
                for (int v = 0; v < 8; ++v)
                    O[(size_t)(mBase + v) * ldo + nCol] = a[v];
            } else {
                __bf16* O = (__bf16*)out;
#pragma unroll
                for (int v = 0; v < 8; ++v)
                    O[(size_t)(mBase + v) * ldo + nCol] = (__bf16)a[v];
            }
        }
    }
}

// ---------------------------------------------------------------------------
// Row softmax over L f32 logits -> bf16 probs (+ optional f32 probs to d_out)
// ---------------------------------------------------------------------------
__global__ void __launch_bounds__(256)
softmax_row(const float* __restrict__ E, __bf16* __restrict__ Abf,
            float* __restrict__ Afp, int L) {
    const int row  = blockIdx.x;
    const int tid  = threadIdx.x;
    const int lane = tid & 31;
    const int wid  = tid >> 5;
    const float* e = E + (size_t)row * L;
    __shared__ float red[8];

    float m = -3.4e38f;
    for (int i = tid; i < L; i += 256) m = fmaxf(m, e[i]);
#pragma unroll
    for (int off = 16; off; off >>= 1) m = fmaxf(m, __shfl_xor(m, off, 32));
    if (lane == 0) red[wid] = m;
    __syncthreads();
    m = red[0];
#pragma unroll
    for (int i = 1; i < 8; ++i) m = fmaxf(m, red[i]);
    __syncthreads();

    float s = 0.0f;
    for (int i = tid; i < L; i += 256) s += __expf(e[i] - m);
#pragma unroll
    for (int off = 16; off; off >>= 1) s += __shfl_xor(s, off, 32);
    if (lane == 0) red[wid] = s;
    __syncthreads();
    s = red[0];
#pragma unroll
    for (int i = 1; i < 8; ++i) s += red[i];
    const float inv = 1.0f / s;

    for (int i = tid; i < L; i += 256) {
        float p = __expf(e[i] - m) * inv;
        Abf[(size_t)row * L + i] = (__bf16)p;
        if (Afp) Afp[(size_t)row * L + i] = p;
    }
}

// ---------------------------------------------------------------------------
// Host orchestration
// ---------------------------------------------------------------------------
extern "C" void kernel_launch(void* const* d_in, const int* in_sizes, int n_in,
                              void* d_out, int out_size, void* d_ws, size_t ws_size,
                              hipStream_t stream) {
    constexpr int B = 4, D = 512, L = 2048, N = 4096;

    const float* H   = (const float*)d_in[0];
    const float* Wk1 = (const float*)d_in[1];
    const float* bk1 = (const float*)d_in[2];
    const float* Wv1 = (const float*)d_in[3];
    const float* bv1 = (const float*)d_in[4];
    const float* Wk2 = (const float*)d_in[5];
    const float* bk2 = (const float*)d_in[6];
    const float* Wv2 = (const float*)d_in[7];
    const float* bv2 = (const float*)d_in[8];
    const float* Qw  = (const float*)d_in[9];

    char*  ws  = (char*)d_ws;
    size_t off = 0;
    auto carve = [&](size_t bytes) -> void* {
        void* p = ws + off;
        off = (off + bytes + 255) & ~(size_t)255;
        return p;
    };
    __bf16* Hbt  = (__bf16*)carve((size_t)B * L * D * 2);  // Ht[b][l][d]
    __bf16* Wk1b = (__bf16*)carve((size_t)D * D * 2);
    __bf16* Wv1b = (__bf16*)carve((size_t)D * D * 2);
    __bf16* Wk2b = (__bf16*)carve((size_t)D * D * 2);
    __bf16* Wv2b = (__bf16*)carve((size_t)D * D * 2);
    __bf16* Qwb  = (__bf16*)carve((size_t)N * D * 2);
    __bf16* Kb1  = (__bf16*)carve((size_t)L * D * 2);      // [l][d]
    __bf16* Vb1  = (__bf16*)carve((size_t)D * L * 2);      // [d][l]
    __bf16* Kb2  = (__bf16*)carve((size_t)L * D * 2);
    __bf16* Vb2  = (__bf16*)carve((size_t)D * L * 2);
    float*  Ebuf = (float*) carve((size_t)N * L * 4);      // per-batch scores
    __bf16* Abuf = (__bf16*)carve((size_t)N * L * 2);      // per-batch probs
    __bf16* C1b  = (__bf16*)carve((size_t)N * D * 2);      // hop-1 context

    // H[b][d][l] f32 -> Ht[b][l][d] bf16
    transpose_f32_bf16<<<dim3(L / 16, D / 16, B), 256, 0, stream>>>(H, Hbt, D, L);
    auto cvt = [&](const float* src, __bf16* dst, int n) {
        cvt_f32_bf16<<<(n + 255) / 256, 256, 0, stream>>>(src, dst, n);
    };
    cvt(Wk1, Wk1b, D * D);
    cvt(Wv1, Wv1b, D * D);
    cvt(Wk2, Wk2b, D * D);
    cvt(Wv2, Wv2b, D * D);
    cvt(Qw,  Qwb,  N * D);

    auto gemm = [&](const __bf16* A, const __bf16* Bt, int M, int Nn, int K,
                    int lda, int ldb, void* out, int ldo,
                    const float* bias, int mode) {
        const int strips = (M / 32) * (Nn / 64);   // all cases divisible by 4
        gemm_wmma_bf16<<<strips / 4, 128, 0, stream>>>(
            A, Bt, Nn / 64, K, lda, ldb, out, ldo, bias, mode);
    };

    float* C2out0 = (float*)d_out;                       // [B][N][D]
    float* A2out0 = (float*)d_out + (size_t)B * N * D;   // [B][N][L]

    for (int b = 0; b < B; ++b) {
        const __bf16* Ht_b = Hbt + (size_t)b * L * D;    // [l][d] bf16

        // projections: ELU(W . H + bias); B-operand = Ht (col-major H)
        gemm(Wk1b, Ht_b, D, L, D, D, D, Kb1, D, bk1, 1); // K1 -> [l][d]
        gemm(Wv1b, Ht_b, D, L, D, D, D, Vb1, L, bv1, 0); // V1 -> [d][l]
        gemm(Wk2b, Ht_b, D, L, D, D, D, Kb2, D, bk2, 1); // K2 -> [l][d]
        gemm(Wv2b, Ht_b, D, L, D, D, D, Vb2, L, bv2, 0); // V2 -> [d][l]

        // hop 1
        gemm(Qwb, Kb1, N, L, D, D, D, Ebuf, L, nullptr, 2);     // E1 = Qw.K1^T
        softmax_row<<<N, 256, 0, stream>>>(Ebuf, Abuf, nullptr, L);
        gemm(Abuf, Vb1, N, D, L, L, L, C1b, D, nullptr, 3);     // C1 = A1.V1

        // hop 2
        gemm(C1b, Kb2, N, L, D, D, D, Ebuf, L, nullptr, 2);     // E2 = C1.K2^T
        float* A2out = A2out0 + (size_t)b * N * L;
        softmax_row<<<N, 256, 0, stream>>>(Ebuf, Abuf, A2out, L);
        float* C2out = C2out0 + (size_t)b * N * D;
        gemm(Abuf, Vb2, N, D, L, L, L, C2out, D, nullptr, 2);   // C2 = A2.V2
    }
}
